// AttnMPNNLayer_25357486915982
// MI455X (gfx1250) — matrix operand order repaired and checked by reading, not compiled, not run on hardware
//
#include <hip/hip_runtime.h>
#include <hip/hip_bf16.h>

#define N_NODES 50000
#define N_EDGES 800000
#define DD 64
#define HH 128
#define KE 192      // 3*D edge-MLP input dim
#define SX 200      // LDS stride (bf16 elems) for edge input rows: 400B, odd multiple of 16B
#define SH 136      // LDS stride for hidden rows: 272B, odd multiple of 16B

typedef __bf16 bf16_t;
typedef __bf16 v16bf __attribute__((ext_vector_type(16)));
typedef __bf16 v8bf  __attribute__((ext_vector_type(8)));
typedef float  v8f   __attribute__((ext_vector_type(8)));

// ---------------- WMMA fragment helpers (wave32, 16x16x32 bf16 -> f32) ----------------

__device__ inline v16bf load_a_lds(const bf16_t* X, int rowBase, int k0, int stride) {
    // 16-bit A 16x32 layout: lanes 0-15 row M=lane, K {0..7,16..23}; lanes 16-31 K {8..15,24..31}
    const int lane = threadIdx.x & 31;
    const int m    = lane & 15;
    const int koff = (lane & 16) ? 8 : 0;
    const bf16_t* p = X + (rowBase + m) * stride + k0 + koff;
    v8bf lo = *(const v8bf*)p;          // K = koff .. koff+7
    v8bf hi = *(const v8bf*)(p + 16);   // K = koff+16 .. koff+23
    v16bf a;
#pragma unroll
    for (int i = 0; i < 8; ++i) { a[i] = lo[i]; a[i + 8] = hi[i]; }
    return a;
}

__device__ inline v16bf load_b_glb(const bf16_t* Wt, int colBase, int k0, int ldk) {
    // B 32x16: lanes 0-15 hold col N=lane%16, K {0..7,16..23}; lanes 16-31 K {8..15,24..31} (2 per VGPR).
    // Wt is stored transposed: [N][K] row-major -> two runs of 8 contiguous bf16 per lane, 16 apart in K.
    const int lane = threadIdx.x & 31;
    const int n    = lane & 15;
    const int krow = (lane & 16) ? 8 : 0;
    const bf16_t* p = Wt + (size_t)(colBase + n) * ldk + k0 + krow;
    v8bf lo = *(const v8bf*)p;
    v8bf hi = *(const v8bf*)(p + 16);
    v16bf b;
#pragma unroll
    for (int i = 0; i < 8; ++i) { b[i] = lo[i]; b[i + 8] = hi[i]; }
    return b;
}

__device__ inline v8f wmma_bf16(v16bf a, v16bf b, v8f c) {
    return __builtin_amdgcn_wmma_f32_16x16x32_bf16(false, a, false, b, (short)0, c, false, false);
}

// epilogue: bias + store (f32 global), optional row guard
__device__ inline void store_c_bias(float* __restrict__ out, int ld, int rowBase, int colBase,
                                    v8f c, float bv, int rowLimit) {
    // f32 C/D 16x16: VGPR r -> M=r (lanes 0-15), M=r+8 (lanes 16-31); N = lane%16
    const int lane = threadIdx.x & 31;
    const int n    = lane & 15;
    const int mb   = (lane & 16) ? 8 : 0;
#pragma unroll
    for (int r = 0; r < 8; ++r) {
        int row = rowBase + mb + r;
        if (row < rowLimit) out[(size_t)row * ld + colBase + n] = c[r] + bv;
    }
}

// epilogue: bias + ReLU + bf16 -> LDS
__device__ inline void store_h_relu_bias(bf16_t* H, int stride, int rowBase, int colBase,
                                         v8f c, float bv) {
    const int lane = threadIdx.x & 31;
    const int n    = lane & 15;
    const int mb   = (lane & 16) ? 8 : 0;
#pragma unroll
    for (int r = 0; r < 8; ++r)
        H[(rowBase + mb + r) * stride + colBase + n] = (bf16_t)fmaxf(c[r] + bv, 0.f);
}

__device__ inline void atomicMaxF(float* addr, float val) {
    unsigned int* ua = (unsigned int*)addr;
    unsigned int cur = *ua;
    while (__uint_as_float(cur) < val) {
        unsigned int prev = atomicCAS(ua, cur, __float_as_uint(val));
        if (prev == cur) break;
        cur = prev;
    }
}

// ---------------- prep kernels ----------------

// dst[n][k] (bf16, [Ndim][Kdim]) = src[k][n] (f32, [Kdim][Ndim])
__global__ __launch_bounds__(256) void transpose_bf16_kernel(bf16_t* __restrict__ dst,
                                                             const float* __restrict__ src,
                                                             int Kdim, int Ndim) {
    int idx = blockIdx.x * 256 + threadIdx.x;
    if (idx < Kdim * Ndim) {
        int n = idx / Kdim;
        int k = idx % Kdim;
        dst[idx] = (bf16_t)src[k * Ndim + n];
    }
}

__global__ __launch_bounds__(256) void init_kernel(float* __restrict__ segmax,
                                                   float* __restrict__ segsum,
                                                   float* __restrict__ cnt,
                                                   float* __restrict__ agg) {
    int i = blockIdx.x * 256 + threadIdx.x;
    if (i < N_NODES) { segmax[i] = -3.402823466e38f; segsum[i] = 0.f; cnt[i] = 0.f; }
    if (i < N_NODES * DD) agg[i] = 0.f;
}

// ---------------- edge kernel: edge MLP + attention MLP + logits ----------------

__global__ __launch_bounds__(256) void edge_kernel(
    const float* __restrict__ nf, const float* __restrict__ ef,
    const bf16_t* __restrict__ We1t, const float* __restrict__ be1,
    const bf16_t* __restrict__ We2t, const float* __restrict__ be2,
    const bf16_t* __restrict__ Wa1t, const float* __restrict__ ba1,
    const float* __restrict__ Wa2,  const float* __restrict__ ba2,
    const int* __restrict__ src, const int* __restrict__ dst,
    float* __restrict__ uef_out, float* __restrict__ logits,
    float* __restrict__ segmax, float* __restrict__ cnt)
{
    __shared__ int sSrc[64], sDst[64];
    __shared__ __align__(16) bf16_t Xs[64 * SX];       // [64][192] edge-MLP input, bf16
    __shared__ __align__(16) bf16_t Hs[2][64 * SH];    // hidden activations: [0]=edge, [1]=attn

    const int t     = threadIdx.x;
    const int eBase = blockIdx.x * 64;

    if (t < 64)            sSrc[t]      = src[eBase + t];
    else if (t < 128)      sDst[t - 64] = dst[eBase + t - 64];

    // L2 prefetch hint for the (shared, hot) transposed weights
    __builtin_prefetch(We1t + (size_t)t * 96, 0, 3);
    __builtin_prefetch(Wa1t + (size_t)t * 96, 0, 3);
    __syncthreads();

    // ---- gather X = [ef | nf[src] | nf[dst]] -> bf16 LDS (float4 granularity) ----
#pragma unroll
    for (int i = 0; i < 12; ++i) {
        int idx = t + i * 256;          // 0..3071 (64 rows * 48 float4)
        int row = idx / 48;
        int c4  = idx % 48;
        int seg = c4 >> 4;              // 0:ef 1:nf[src] 2:nf[dst]
        int o   = (c4 & 15) << 2;       // float offset within 64-wide segment
        const float* p;
        if (seg == 0)      p = ef + (size_t)(eBase + row) * DD + o;
        else if (seg == 1) p = nf + (size_t)sSrc[row] * DD + o;
        else               p = nf + (size_t)sDst[row] * DD + o;
        float4 v = *(const float4*)p;
        bf16_t* q = &Xs[row * SX + seg * DD + o];
        q[0] = (bf16_t)v.x; q[1] = (bf16_t)v.y; q[2] = (bf16_t)v.z; q[3] = (bf16_t)v.w;
    }
    __syncthreads();

    const int wave = t >> 5;
    const int nIdx = t & 15;

    // ---- GEMM1 (x2 MLPs): [64x192] @ [192x128], bias + ReLU -> bf16 LDS ----
    // wave -> (mlp, rowT); A fragments loaded once, resident in VGPRs for all 8 colT tiles
    {
        const int mlp  = wave >> 2;                  // waves 0-3: edge MLP, 4-7: attn MLP
        const int rowT = (wave & 3) << 4;
        const bf16_t* W   = mlp ? Wa1t : We1t;
        const float* bias = mlp ? ba1  : be1;
        bf16_t* Hrow = &Hs[mlp][0];

        v16bf aF[6];
#pragma unroll
        for (int kk = 0; kk < 6; ++kk) aF[kk] = load_a_lds(Xs, rowT, kk * 32, SX);

#pragma unroll
        for (int c8 = 0; c8 < 8; ++c8) {
            int colT = c8 << 4;
            v8f acc;
#pragma unroll
            for (int r = 0; r < 8; ++r) acc[r] = 0.f;
#pragma unroll
            for (int kk = 0; kk < 6; ++kk)
                acc = wmma_bf16(aF[kk], load_b_glb(W, colT, kk * 32, KE), acc);
            store_h_relu_bias(Hrow, SH, rowT, colT, acc, bias[colT + nIdx]);
        }
    }
    __syncthreads();

    // ---- GEMM2: He [64x128] @ We2t [128x64] + be2 -> uef (f32, global) ----
    // wave -> (rowT, colT pair); 4 resident A fragments
    {
        const int rowT = (wave & 3) << 4;
        const int cp   = (wave >> 2) << 5;           // 0 or 32

        v16bf aF[4];
#pragma unroll
        for (int kk = 0; kk < 4; ++kk) aF[kk] = load_a_lds(&Hs[0][0], rowT, kk * 32, SH);

#pragma unroll
        for (int j = 0; j < 2; ++j) {
            int colT = cp + (j << 4);
            v8f acc;
#pragma unroll
            for (int r = 0; r < 8; ++r) acc[r] = 0.f;
#pragma unroll
            for (int kk = 0; kk < 4; ++kk)
                acc = wmma_bf16(aF[kk], load_b_glb(We2t, colT, kk * 32, HH), acc);
            store_c_bias(uef_out, DD, eBase + rowT, colT, acc, be2[colT + nIdx], N_EDGES);
        }
    }

    // ---- logits: Ha [64x128] @ Wa2 [128] + ba2; atomic seg-max & count ----
    {
        int eL   = t >> 2;                  // 4 threads per edge
        int part = t & 3;
        const bf16_t* hr = &Hs[1][eL * SH + part * 32];
        float s = 0.f;
#pragma unroll
        for (int i = 0; i < 32; ++i) s += (float)hr[i] * Wa2[part * 32 + i];
        s += __shfl_xor(s, 1);
        s += __shfl_xor(s, 2);
        if (part == 0) {
            float lg = s + ba2[0];
            logits[eBase + eL] = lg;
            int dnode = sDst[eL];
            atomicMaxF(&segmax[dnode], lg);
            atomicAdd(&cnt[dnode], 1.0f);
        }
    }
}

// ---------------- softmax numerator + segment sum ----------------

__global__ __launch_bounds__(256) void exp_kernel(
    const float* __restrict__ logits, const float* __restrict__ segmax,
    const int* __restrict__ dst, float* __restrict__ expw, float* __restrict__ segsum)
{
    int e = blockIdx.x * 256 + threadIdx.x;
    if (e < N_EDGES) {
        int d = dst[e];
        float ex = __expf(logits[e] - segmax[d]);
        expw[e] = ex;
        atomicAdd(&segsum[d], ex);
    }
}

// ---------------- attention-weighted message scatter-add ----------------

__global__ __launch_bounds__(256) void scatter_kernel(
    const float* __restrict__ uef, const float* __restrict__ expw,
    const float* __restrict__ segsum, const int* __restrict__ dst,
    float* __restrict__ agg)
{
    int idx = blockIdx.x * 256 + threadIdx.x;
    if (idx < N_EDGES * DD) {
        int e = idx >> 6;
        int c = idx & 63;
        int d = dst[e];
        float attn = expw[e] / segsum[d];
        atomicAdd(&agg[d * DD + c], uef[(size_t)e * DD + c] * attn);
    }
}

// ---------------- node kernel: [agg_m | nf] MLP ----------------

__global__ __launch_bounds__(256) void node_kernel(
    const float* __restrict__ nf, const float* __restrict__ agg,
    const float* __restrict__ cnt,
    const bf16_t* __restrict__ Wn1t, const float* __restrict__ bn1,
    const bf16_t* __restrict__ Wn2t, const float* __restrict__ bn2,
    float* __restrict__ unf_out)
{
    __shared__ __align__(16) bf16_t X2[64 * SH];   // [64][128] node-MLP input
    __shared__ __align__(16) bf16_t Hn[64 * SH];   // [64][128] hidden

    const int t     = threadIdx.x;
    const int nBase = blockIdx.x * 64;

#pragma unroll
    for (int i = 0; i < 8; ++i) {
        int idx = t + i * 256;          // 0..2047 (64 rows * 32 float4)
        int row = idx >> 5;
        int c4  = idx & 31;
        int seg = c4 >> 4;              // 0: agg_m, 1: nf
        int o   = (c4 & 15) << 2;
        int n   = nBase + row;
        int nc  = (n < N_NODES) ? n : 0;
        float4 v;
        float sc = 1.f;
        if (seg == 0) {
            v  = *(const float4*)(agg + (size_t)nc * DD + o);
            sc = 1.f / fmaxf(cnt[nc], 1.f);
        } else {
            v  = *(const float4*)(nf + (size_t)nc * DD + o);
        }
        bf16_t* q = &X2[row * SH + seg * DD + o];
        q[0] = (bf16_t)(v.x * sc); q[1] = (bf16_t)(v.y * sc);
        q[2] = (bf16_t)(v.z * sc); q[3] = (bf16_t)(v.w * sc);
    }
    __syncthreads();

    const int wave = t >> 5;
    const int nIdx = t & 15;

    // ---- GEMM1: X2 [64x128] @ Wn1t [128x128] + bn1, ReLU -> Hn ----
    // wave -> (rowT, colT quad); 4 resident A fragments, 4 colT tiles each
    {
        const int rowT = (wave & 3) << 4;
        const int cq   = (wave >> 2) << 6;           // 0 or 64

        v16bf aF[4];
#pragma unroll
        for (int kk = 0; kk < 4; ++kk) aF[kk] = load_a_lds(X2, rowT, kk * 32, SH);

#pragma unroll
        for (int j = 0; j < 4; ++j) {
            int colT = cq + (j << 4);
            v8f acc;
#pragma unroll
            for (int r = 0; r < 8; ++r) acc[r] = 0.f;
#pragma unroll
            for (int kk = 0; kk < 4; ++kk)
                acc = wmma_bf16(aF[kk], load_b_glb(Wn1t, colT, kk * 32, HH), acc);
            store_h_relu_bias(Hn, SH, rowT, colT, acc, bn1[colT + nIdx]);
        }
    }
    __syncthreads();

    // ---- GEMM2: Hn [64x128] @ Wn2t [128x64] + bn2 -> unf (f32, global, guarded) ----
    {
        const int rowT = (wave & 3) << 4;
        const int cp   = (wave >> 2) << 5;           // 0 or 32

        v16bf aF[4];
#pragma unroll
        for (int kk = 0; kk < 4; ++kk) aF[kk] = load_a_lds(Hn, rowT, kk * 32, SH);

#pragma unroll
        for (int j = 0; j < 2; ++j) {
            int colT = cp + (j << 4);
            v8f acc;
#pragma unroll
            for (int r = 0; r < 8; ++r) acc[r] = 0.f;
#pragma unroll
            for (int kk = 0; kk < 4; ++kk)
                acc = wmma_bf16(aF[kk], load_b_glb(Wn2t, colT, kk * 32, HH), acc);
            store_c_bias(unf_out, DD, nBase + rowT, colT, acc, bn2[colT + nIdx], N_NODES);
        }
    }
}

// ---------------- launcher ----------------

extern "C" void kernel_launch(void* const* d_in, const int* in_sizes, int n_in,
                              void* d_out, int out_size, void* d_ws, size_t ws_size,
                              hipStream_t stream) {
    const float* nf  = (const float*)d_in[0];
    const float* ef  = (const float*)d_in[1];
    const float* We1 = (const float*)d_in[2];
    const float* be1 = (const float*)d_in[3];
    const float* We2 = (const float*)d_in[4];
    const float* be2 = (const float*)d_in[5];
    const float* Wa1 = (const float*)d_in[6];
    const float* ba1 = (const float*)d_in[7];
    const float* Wa2 = (const float*)d_in[8];
    const float* ba2 = (const float*)d_in[9];
    const float* Wn1 = (const float*)d_in[10];
    const float* bn1 = (const float*)d_in[11];
    const float* Wn2 = (const float*)d_in[12];
    const float* bn2 = (const float*)d_in[13];
    const int*   src = (const int*)d_in[14];
    const int*   dst = (const int*)d_in[15];

    float* unf_out = (float*)d_out;                       // [N, 64] first
    float* uef_out = unf_out + (size_t)N_NODES * DD;      // [E, 64] second

    char* ws = (char*)d_ws;
    size_t off = 0;
    auto take = [&](size_t b) { char* p = ws + off; off = (off + b + 255) & ~(size_t)255; return p; };
    bf16_t* we1t   = (bf16_t*)take((size_t)HH * KE * 2);
    bf16_t* wa1t   = (bf16_t*)take((size_t)HH * KE * 2);
    bf16_t* we2t   = (bf16_t*)take((size_t)DD * HH * 2);
    bf16_t* wn1t   = (bf16_t*)take((size_t)HH * HH * 2);
    bf16_t* wn2t   = (bf16_t*)take((size_t)DD * HH * 2);
    float*  segmax = (float*)take((size_t)N_NODES * 4);
    float*  segsum = (float*)take((size_t)N_NODES * 4);
    float*  cnt    = (float*)take((size_t)N_NODES * 4);
    float*  logits = (float*)take((size_t)N_EDGES * 4);
    float*  expw   = (float*)take((size_t)N_EDGES * 4);
    float*  agg    = (float*)take((size_t)N_NODES * DD * 4);
    (void)in_sizes; (void)n_in; (void)out_size; (void)ws_size;

    // weight prep (transposed bf16) + per-launch state init
    transpose_bf16_kernel<<<(HH * KE + 255) / 256, 256, 0, stream>>>(we1t, We1, KE, HH);
    transpose_bf16_kernel<<<(HH * KE + 255) / 256, 256, 0, stream>>>(wa1t, Wa1, KE, HH);
    transpose_bf16_kernel<<<(DD * HH + 255) / 256, 256, 0, stream>>>(we2t, We2, HH, DD);
    transpose_bf16_kernel<<<(HH * HH + 255) / 256, 256, 0, stream>>>(wn1t, Wn1, HH, HH);
    transpose_bf16_kernel<<<(DD * HH + 255) / 256, 256, 0, stream>>>(wn2t, Wn2, HH, DD);
    init_kernel<<<(N_NODES * DD + 255) / 256, 256, 0, stream>>>(segmax, segsum, cnt, agg);

    // edge MLPs + logits (+ seg-max, counts)
    edge_kernel<<<N_EDGES / 64, 256, 0, stream>>>(
        nf, ef, we1t, be1, we2t, be2, wa1t, ba1, Wa2, ba2,
        src, dst, uef_out, logits, segmax, cnt);

    // softmax numerator + denominator
    exp_kernel<<<(N_EDGES + 255) / 256, 256, 0, stream>>>(logits, segmax, dst, expw, segsum);

    // attention-weighted message aggregation
    scatter_kernel<<<(N_EDGES * DD + 255) / 256, 256, 0, stream>>>(uef_out, expw, segsum, dst, agg);

    // node MLP
    node_kernel<<<(N_NODES + 63) / 64, 256, 0, stream>>>(
        nf, agg, cnt, wn1t, bn1, wn2t, bn2, unf_out);
}
